// Dynamic_conv2d_49795850830311
// MI455X (gfx1250) — matrix operand order repaired and checked by reading, not compiled, not run on hardware
//
#include <hip/hip_runtime.h>

// ---------------- problem constants ----------------
#define B_      32
#define C_      128
#define H_      56
#define W_      56
#define O_      128
#define HW_     (H_ * W_)        // 3136
#define KDIM_   (C_ * 9)         // 1152, K order: k' = r*128 + c (tap-major)
#define NSTEP_  (KDIM_ / 32)     // 36 K-steps of 32 (4 steps per tap r)
#define NTILE_  64               // spatial columns per block
#define NBLK_   (HW_ / NTILE_)   // 49

#define GSTRIDE 36               // f32 gather tile row stride (floats)
#define CSTRIDE 40               // bf16 tile row stride (bf16 elems) -> 80B, bank-friendly
#define ESTRIDE 68               // epilogue tile row stride (floats) -> 272B, 16B aligned

typedef __attribute__((ext_vector_type(16))) __bf16 bf16x16;
typedef __attribute__((ext_vector_type(2)))  __bf16 bf16x2;
typedef __attribute__((ext_vector_type(8)))  float  f32x8;

__device__ __forceinline__ unsigned short f32_bf16(float f) {
  unsigned u = __float_as_uint(f);
  u += 0x7FFFu + ((u >> 16) & 1u);          // round-to-nearest-even
  return (unsigned short)(u >> 16);
}

// ---------------- kernel 0: global average pool -> pooled[B*C] ----------------
__global__ __launch_bounds__(256) void k_pool(const float* __restrict__ x,
                                              float* __restrict__ pooled) {
  __shared__ float red[256];
  const int bc = blockIdx.x;
  const float* p = x + (size_t)bc * HW_;
  float s = 0.f;
  for (int i = threadIdx.x; i < HW_; i += 256) s += p[i];
  red[threadIdx.x] = s;
  __syncthreads();
  for (int off = 128; off > 0; off >>= 1) {
    if (threadIdx.x < (unsigned)off) red[threadIdx.x] += red[threadIdx.x + off];
    __syncthreads();
  }
  if (threadIdx.x == 0) pooled[bc] = red[0] * (1.0f / (float)HW_);
}

// ---------------- kernel 1: attention (fc1->relu->fc2->softmax/T) + agg bias ----------------
__global__ __launch_bounds__(256) void k_attn(const float* __restrict__ pooled,
                                              const float* __restrict__ fc1,   // [32][128]
                                              const float* __restrict__ fc2,   // [2][32]
                                              const float* __restrict__ cb,    // [128]
                                              const float* __restrict__ db,    // [128]
                                              float* __restrict__ attn_out,    // [B][2]
                                              float* __restrict__ aggb) {      // [B][128]
  __shared__ float pl[B_ * C_];
  __shared__ float hid[B_ * 32];
  __shared__ float att[B_ * 2];
  const int t = threadIdx.x;
  for (int i = t; i < B_ * C_; i += 256) pl[i] = pooled[i];
  __syncthreads();
  for (int id = t; id < B_ * 32; id += 256) {
    const int b = id >> 5, h = id & 31;
    float s = 0.f;
    for (int c = 0; c < C_; ++c) s += pl[b * C_ + c] * fc1[h * C_ + c];
    hid[id] = s > 0.f ? s : 0.f;
  }
  __syncthreads();
  if (t < B_) {
    float l0 = 0.f, l1 = 0.f;
    for (int h = 0; h < 32; ++h) {
      const float v = hid[t * 32 + h];
      l0 += v * fc2[h];
      l1 += v * fc2[32 + h];
    }
    l0 *= (1.0f / 31.0f); l1 *= (1.0f / 31.0f);
    const float m = fmaxf(l0, l1);
    const float e0 = __expf(l0 - m), e1 = __expf(l1 - m);
    const float inv = 1.0f / (e0 + e1);
    att[t * 2 + 0] = e0 * inv;  att[t * 2 + 1] = e1 * inv;
    attn_out[t * 2 + 0] = e0 * inv;  attn_out[t * 2 + 1] = e1 * inv;
  }
  __syncthreads();
  for (int i = t; i < B_ * O_; i += 256) {
    const int b = i >> 7, o = i & 127;
    aggb[i] = att[b * 2] * cb[o] + att[b * 2 + 1] * db[o];
  }
}

// ---------------- kernel 2: aggregated weights -> bf16 [b][o][k'], k' = r*128 + c ----------------
__global__ __launch_bounds__(256) void k_wagg(const float* __restrict__ cw,
                                              const float* __restrict__ dw,
                                              const float* __restrict__ attn,
                                              unsigned short* __restrict__ wbf) {
  const int idx = blockIdx.x * 256 + threadIdx.x;      // exact: B*O*KDIM elements
  const int per = O_ * KDIM_;                          // 147456
  const int b = idx / per;
  int rem = idx - b * per;
  const int o = rem / KDIM_;
  const int kp = rem - o * KDIM_;                      // k' = r*128 + c
  const int r = kp >> 7;                               // tap 0..8
  const int c = kp & 127;                              // channel
  const int src = (o * C_ + c) * 9 + r;                // source [o][c][3][3]
  const float a0 = attn[b * 2], a1 = attn[b * 2 + 1];
  wbf[idx] = f32_bf16(a0 * cw[src] + a1 * dw[src]);
}

// ---------------- async im2col staging (tap-major K): 32(K) x 64(N) f32 tile ----------------
// Step ks covers tap r = ks>>2 (wave-uniform), channels c = (ks&3)*32 + kk.
// x offset = c*HW + n + (ky-1)*56 + (kx-1): no per-element div/mod.
__device__ __forceinline__ void stage_tile(const float* __restrict__ xb,
                                           float* buf, float* dummy,
                                           int t, int lane, int nn, int oy, int ox,
                                           int n, int ks) {
  const int c0 = (ks & 3) * 32;
  const int r  = ks >> 2;
  const int ky = r / 3, kx = r - ky * 3;           // uniform scalars
  const int iy = oy + ky - 1, ix = ox + kx - 1;
  const bool ok = ((unsigned)iy < 56u) & ((unsigned)ix < 56u);
  const int kkb = t >> 6;                          // 0..3
  const float* gb = xb + (size_t)(c0 + kkb) * HW_ + (n + (ky - 1) * 56 + (kx - 1));
#pragma unroll
  for (int e = 0; e < 8; ++e) {
    const int kk = e * 4 + kkb;                    // 0..31
    const float* g = ok ? (gb + e * 4 * HW_) : xb;           // safe addr for pad lanes
    float* l = ok ? (buf + nn * GSTRIDE + kk) : (dummy + lane);
    const unsigned loff = (unsigned)(unsigned long long)l;   // LDS byte offset
    // CDNA5 async DMA gather: global -> LDS, ASYNCcnt-tracked, full EXEC (branchless)
    asm volatile("global_load_async_to_lds_b32 %0, %1, off"
                 :: "v"(loff), "v"(g) : "memory");
  }
  if (!ok) {                                       // one EXEC toggle, 8 zero ds_stores
#pragma unroll
    for (int e = 0; e < 8; ++e) buf[nn * GSTRIDE + e * 4 + kkb] = 0.0f;
  }
}

// ---------------- kernel 3: implicit-GEMM conv, WMMA bf16 ----------------
// D = A x B with A = im2col (M=16 spatial), B = weights (N=16 output chans).
__global__ __launch_bounds__(256) void k_conv(const float* __restrict__ x,
                                              const unsigned short* __restrict__ wbf, // [B][O][KDIM] bf16
                                              const float* __restrict__ aggb,         // [B][O]
                                              float* __restrict__ out) {              // [B][O][HW]
  __shared__ __align__(16) float gbuf[2][NTILE_ * GSTRIDE];        // 2 x 9216 B
  __shared__ __align__(16) unsigned short cbuf[NTILE_ * CSTRIDE];  // 5120 B
  __shared__ float dummy[32];
  __shared__ __align__(16) float ebuf[8 * 16 * ESTRIDE];           // 34816 B

  const int t = threadIdx.x;
  const int lane = t & 31;
  const int wave = t >> 5;                   // 8 waves -> N tiles of 16 over O=128
  const int blk = blockIdx.x;
  const int b = blk / NBLK_;
  const int nb = (blk - b * NBLK_) * NTILE_;
  const float* xb = x + (size_t)b * C_ * HW_;

  const int row = lane & 15;
  const int hi  = lane >> 4;

  // per-thread gather coordinates (constant across all K-steps)
  const int nn = t & 63;
  const int n  = nb + nn;
  const int oy = n / 56, ox = n - oy * 56;

  // weight (B operand) pointer: lane -> output channel column, hi -> K half
  const unsigned short* bp =
      wbf + ((size_t)(b * O_ + wave * 16 + row)) * KDIM_ + hi * 16;

  f32x8 acc[4] = {};                         // 64(n) x 16(o) per wave

  stage_tile(xb, gbuf[0], dummy, t, lane, nn, oy, ox, n, 0);

  for (int ks = 0; ks < NSTEP_; ++ks) {
    float* gcur = gbuf[ks & 1];
    if (ks + 1 < NSTEP_) {
      stage_tile(xb, gbuf[(ks + 1) & 1], dummy, t, lane, nn, oy, ox, n, ks + 1);
      // exactly 8 async issues per wave per stage -> <=8 leaves only the prefetch in flight
      asm volatile("s_wait_asynccnt 0x8" ::: "memory");
      __builtin_prefetch((const void*)(bp + (ks + 1) * 32), 0, 0);  // global_prefetch_b8
    } else {
      asm volatile("s_wait_asynccnt 0x0" ::: "memory");
    }
    __syncthreads();                          // gather tile for ks visible

    // shared conversion pass: f32 [n][k] -> bf16 [n][k], once per block, native cvt
#pragma unroll
    for (int e = 0; e < 4; ++e) {
      const int p = e * 256 + t;              // 0..1023 k-pairs
      const int kk2 = p >> 6;                 // 0..15
      const int n2 = p & 63;
      const float2 f = *(const float2*)(gcur + n2 * GSTRIDE + kk2 * 2);
      bf16x2 pk = { (__bf16)f.x, (__bf16)f.y };
      *(bf16x2*)(cbuf + n2 * CSTRIDE + kk2 * 2) = pk;
    }
    __syncthreads();                          // bf16 tile ready

    // B fragment (32x16 bf16 weights): lanes 0-15 K 0-15 of col, 16-31 K 16-31
    union { uint4 q[2]; bf16x16 v; } wf;
    const unsigned short* wp = bp + ks * 32;
    wf.q[0] = *(const uint4*)(wp);
    wf.q[1] = *(const uint4*)(wp + 8);

#pragma unroll
    for (int sub = 0; sub < 4; ++sub) {
      // A fragment (16x32 bf16 im2col): lanes 0-15 K 0-15, lanes 16-31 K 16-31
      union { uint4 q[2]; bf16x16 v; } af;
      const unsigned short* ar = cbuf + (sub * 16 + row) * CSTRIDE + hi * 16;
      af.q[0] = *(const uint4*)(ar);          // ds_load_b128
      af.q[1] = *(const uint4*)(ar + 8);      // ds_load_b128
      acc[sub] = __builtin_amdgcn_wmma_f32_16x16x32_bf16(
          false, af.v, false, wf.v, (short)0, acc[sub], false, false);
    }
    __syncthreads();                          // free cbuf / gcur for next iteration
  }

  // ---------------- epilogue: bias + LDS transpose -> coalesced b128 stores ----------------
  // D layout: VGPR j, lanes 0-15: M=j (spatial), N=lane (o); lanes 16-31: M=j+8, N=lane-16
  const float bias = aggb[b * O_ + wave * 16 + row];
  float* eb = &ebuf[wave * 16 * ESTRIDE];    // per-wave 16(o) x 64(n)
#pragma unroll
  for (int sub = 0; sub < 4; ++sub)
#pragma unroll
    for (int j = 0; j < 8; ++j)
      eb[row * ESTRIDE + sub * 16 + hi * 8 + j] = acc[sub][j] + bias;
  // per-wave DS ops are in-order; read back row-major and store coalesced
#pragma unroll
  for (int it = 0; it < 8; ++it) {
    const int idx = it * 32 + lane;
    const int orow = idx >> 4, nq = idx & 15;
    const float4 v = *(const float4*)(&eb[orow * ESTRIDE + nq * 4]);
    *(float4*)(out + ((size_t)(b * O_ + wave * 16 + orow)) * HW_ + nb + nq * 4) = v;
  }
}

// ---------------- host-side launch ----------------
extern "C" void kernel_launch(void* const* d_in, const int* in_sizes, int n_in,
                              void* d_out, int out_size, void* d_ws, size_t ws_size,
                              hipStream_t stream) {
  (void)in_sizes; (void)n_in; (void)out_size; (void)ws_size;
  const float* x   = (const float*)d_in[0];
  const float* cw  = (const float*)d_in[1];
  const float* cb  = (const float*)d_in[2];
  const float* dw  = (const float*)d_in[3];
  const float* db  = (const float*)d_in[4];
  const float* fc1 = (const float*)d_in[5];
  const float* fc2 = (const float*)d_in[6];
  float* out = (float*)d_out;

  // workspace layout (~9.47 MB total)
  char* ws = (char*)d_ws;
  unsigned short* wbf = (unsigned short*)ws;                 // B*O*KDIM bf16 = 9,437,184 B
  float* pooled = (float*)(ws + 9437184);                    // B*C f32 = 16 KB
  float* attn   = (float*)(ws + 9437184 + 16384);            // B*2 f32
  float* aggb   = (float*)(ws + 9437184 + 16384 + 256);      // B*O f32 = 16 KB

  k_pool<<<dim3(B_ * C_), dim3(256), 0, stream>>>(x, pooled);
  k_attn<<<dim3(1), dim3(256), 0, stream>>>(pooled, fc1, fc2, cb, db, attn, aggb);
  k_wagg<<<dim3((B_ * O_ * KDIM_) / 256), dim3(256), 0, stream>>>(cw, dw, attn, wbf);
  k_conv<<<dim3(B_ * NBLK_), dim3(256), 0, stream>>>(x, wbf, aggb, out);
}
